// BlockMaskGenerator_69973607186866
// MI455X (gfx1250) — compile-verified, matrix operand order checked by reading, not run on hardware
//
#include <hip/hip_runtime.h>
#include <hip/hip_bf16.h>

typedef __attribute__((ext_vector_type(16))) _Float16 v16h;
typedef __attribute__((ext_vector_type(8)))  float    v8f;

#define NUM_BLOCKS 4
#define SCALE_MIN  0.15f
#define SCALE_MAX  0.20f
#define H 64
#define W 64
#define SEQ (H * W)
#define ROWS_PER_WG 8   // 8 waves of 32 -> 256 threads, one batch row per wave

__global__ __launch_bounds__(256) void BlockMaskGenerator_kernel(
    const float* __restrict__ scales,
    const float* __restrict__ rtops,
    const float* __restrict__ rlefts,
    float* __restrict__ ctx,
    float* __restrict__ tgt,
    float* __restrict__ pos,
    float* __restrict__ cnt,
    int batch)
{
    const int lane = threadIdx.x & 31;
    const int wave = threadIdx.x >> 5;
    const int b    = blockIdx.x * ROWS_PER_WG + wave;
    if (b >= batch) return;   // wave-uniform; EXEC stays all-ones for WMMA

    // ---- per-rectangle params (uniform per wave; same-address broadcast loads)
    int top[4], bh[4], left[4], bw[4];
    unsigned long long colmask[4];
#pragma unroll
    for (int k = 0; k < 4; ++k) {
        int i = b * NUM_BLOCKS + k;
        float s  = SCALE_MIN + scales[i] * (SCALE_MAX - SCALE_MIN);
        int area = (int)(s * (float)(H * W));                 // trunc like .astype(int32)
        int h = (int)sqrtf((float)area);
        h = min(max(h, 1), H);
        int w = (int)((float)area / (float)h);
        w = min(max(w, 1), W);
        int mt = max(H - h + 1, 1);
        int ml = max(W - w + 1, 1);
        top[k]  = (int)(rtops[i]  * (float)mt);
        left[k] = (int)(rlefts[i] * (float)ml);
        bh[k] = h; bw[k] = w;
        colmask[k] = (w >= 64) ? ~0ull
                               : ((((unsigned long long)1 << w) - 1ull) << left[k]);
    }

    // ---- WMMA fragments: tile = A(16x4 row-indicator) x B(4x16 col-indicator)
    // v_wmma_f32_16x16x32_f16 with only K=0..3 populated (exact: 0/1 values, sum<=4).
    // A 16x32 f16 layout: lanes 0-15 carry M=lane, elems 0..3 = K=0..3; lanes 16-31 = K>=8 -> 0.
    // B 32x16 f16 layout: lanes 0-15 carry N=lane, elems 0..3 = K=0..3; lanes 16-31 = K>=8 -> 0.
    v16h afrag[4], bfrag[4];
#pragma unroll
    for (int t = 0; t < 4; ++t) {
        v16h a = {};
        v16h bb = {};
        if (lane < 16) {
            int q = t * 16 + lane;     // row coord for A, col coord for B
#pragma unroll
            for (int k = 0; k < 4; ++k) {
                a[k]  = (q >= top[k]  && q < top[k]  + bh[k]) ? (_Float16)1.0f : (_Float16)0.0f;
                bb[k] = (q >= left[k] && q < left[k] + bw[k]) ? (_Float16)1.0f : (_Float16)0.0f;
            }
        }
        afrag[t] = a; bfrag[t] = bb;
    }

    // ---- 16 WMMAs cover the 64x64 mask; stream target/context with NT stores.
    // D holds exact integer counts 0..4: med3(d,0,1) == (d>0) in ONE v_med3_f32.
    // Compute all 16 values per tile into distinct regs BEFORE storing so no
    // store data register is overwritten while the store is in flight (avoids
    // per-store s_wait_xcnt serialization seen in the previous build).
    const long long base = (long long)b * SEQ;
    const int n     = lane & 15;
    const int mbase = (lane < 16) ? 0 : 8;   // D layout: lanes>=16 hold M=v+8, N=lane-16
#pragma unroll
    for (int tr = 0; tr < 4; ++tr) {
#pragma unroll
        for (int tc = 0; tc < 4; ++tc) {
            v8f c0 = {};
            v8f d = __builtin_amdgcn_wmma_f32_16x16x32_f16(
                false, afrag[tr], false, bfrag[tc], (short)0, c0, false, false);

            float tv[8], cv[8];
#pragma unroll
            for (int v = 0; v < 8; ++v) {
                tv[v] = __builtin_amdgcn_fmed3f(d[v], 0.0f, 1.0f);  // exact 0.0/1.0
                cv[v] = 1.0f - tv[v];
            }
            const long long o0 = base + (long long)(tr * 16 + mbase) * W + (tc * 16 + n);
#pragma unroll
            for (int v = 0; v < 8; ++v)
                __builtin_nontemporal_store(tv[v], tgt + o0 + v * W);
#pragma unroll
            for (int v = 0; v < 8; ++v)
                __builtin_nontemporal_store(cv[v], ctx + o0 + v * W);
        }
    }

    // ---- stable partition (argsort(~mask, stable)): ones first ascending, then zeros.
    // Each lane owns 2 image rows as 64-bit masks; wave scan gives ranks.
    int r0 = 2 * lane, r1 = 2 * lane + 1;
    unsigned long long bits0 = 0, bits1 = 0;
#pragma unroll
    for (int k = 0; k < 4; ++k) {
        if (r0 >= top[k] && r0 < top[k] + bh[k]) bits0 |= colmask[k];
        if (r1 >= top[k] && r1 < top[k] + bh[k]) bits1 |= colmask[k];
    }
    int cl   = __popcll(bits0) + __popcll(bits1);
    int incl = cl;
#pragma unroll
    for (int off = 1; off < 32; off <<= 1) {      // wave32 inclusive scan
        int v = __shfl_up(incl, off, 32);
        if (lane >= off) incl += v;
    }
    int excl  = incl - cl;
    int total = __shfl(incl, 31, 32);
    if (lane == 0) cnt[b] = (float)total;

    // branchless stable scatter: one NT store per element, cndmask-selected offset
    float* posb  = pos + base;
    int oneRank  = excl;
    int zeroRank = total + lane * 128 - excl;     // zeros region base + zeros before lane
    unsigned long long bits = bits0;
    int rbase = r0 * 64;
#pragma unroll 1
    for (int half = 0; half < 2; ++half) {
#pragma unroll 1
        for (int c2 = 0; c2 < 64; ++c2) {
            int bit = (int)((bits >> c2) & 1ull);
            int off = bit ? oneRank : zeroRank;
            __builtin_nontemporal_store((float)(rbase + c2), posb + off);
            oneRank  += bit;
            zeroRank += 1 - bit;
        }
        bits = bits1; rbase = r1 * 64;
    }
}

extern "C" void kernel_launch(void* const* d_in, const int* in_sizes, int n_in,
                              void* d_out, int out_size, void* d_ws, size_t ws_size,
                              hipStream_t stream) {
    const float* scales = (const float*)d_in[0];
    const float* rtops  = (const float*)d_in[1];
    const float* rlefts = (const float*)d_in[2];
    int batch = in_sizes[0] / NUM_BLOCKS;         // TB = B * NUM_BLOCKS

    float* out = (float*)d_out;                   // concat: ctx | tgt | pos | counts
    float* ctx = out;
    float* tgt = out + (long long)batch * SEQ;
    float* pos = out + 2LL * (long long)batch * SEQ;
    float* cnt = out + 3LL * (long long)batch * SEQ;

    int wgs = (batch + ROWS_PER_WG - 1) / ROWS_PER_WG;
    BlockMaskGenerator_kernel<<<wgs, 256, 0, stream>>>(
        scales, rtops, rlefts, ctx, tgt, pos, cnt, batch);
}
